// DecoderLayer_7275674600369
// MI455X (gfx1250) — matrix operand order repaired and checked by reading, not compiled, not run on hardware
//
#include <hip/hip_runtime.h>

#define EMB 1024
#define NHEAD 16
#define DKH 64
#define DFF 4096
#define BATCH 2
#define SEQ 2048
#define TOK (BATCH*SEQ)
#define LN_EPS 1e-5f

typedef __attribute__((ext_vector_type(16))) __bf16 bf16x16;
typedef __attribute__((ext_vector_type(8)))  float  f32x8;
typedef __attribute__((ext_vector_type(4)))  int    v4i;

union FragAB { bf16x16 v; unsigned short u[16]; uint4 q[2]; };
union FragC  { f32x8 v; float f[8]; };

__device__ __forceinline__ unsigned short f2bf(float f) {
    union { float f; unsigned int u; } c; c.f = f;
    unsigned int u = c.u;
    unsigned int r = u + 0x7fffu + ((u >> 16) & 1u);   // round-to-nearest-even
    return (unsigned short)(r >> 16);
}

// ---------------- CDNA5 async global->LDS staging (guarded) ------------------
#if __has_builtin(__builtin_amdgcn_global_load_async_to_lds_b128)
#define HAVE_ASYNC 1
#else
#define HAVE_ASYNC 0
#endif

#if HAVE_ASYNC
typedef __attribute__((address_space(1))) v4i* g128_t;   // "__device__" ptr
typedef __attribute__((address_space(3))) v4i* l128_t;   // "__shared__" ptr
__device__ __forceinline__ void async_copy16(const void* g, const void* l) {
    __builtin_amdgcn_global_load_async_to_lds_b128(
        (g128_t)(unsigned long long)g,
        (l128_t)(unsigned int)(unsigned long long)l,
        0, 0);
}
template<int N>
__device__ __forceinline__ void wait_async() {
#if __has_builtin(__builtin_amdgcn_s_wait_asynccnt)
    __builtin_amdgcn_s_wait_asynccnt(N);
#else
    asm volatile("s_wait_asynccnt %0" :: "i"(N) : "memory");
#endif
}
#endif

// ---------------------------------------------------------------------------
// Weight convert + transpose: W fp32 [K][N] -> Wt bf16 [N][K]
// ---------------------------------------------------------------------------
__global__ __launch_bounds__(256) void wconv_t(
    const float* __restrict__ W, unsigned short* __restrict__ Wt, int K, int N)
{
    __shared__ unsigned short t[64][65];
    const int k0 = blockIdx.x * 64, n0 = blockIdx.y * 64;
    const int tx = threadIdx.x & 63, ty = threadIdx.x >> 6;
#pragma unroll
    for (int i = 0; i < 16; i++) {
        int r = ty + i * 4;
        t[r][tx] = f2bf(W[(size_t)(k0 + r) * N + n0 + tx]);
    }
    __syncthreads();
#pragma unroll
    for (int i = 0; i < 16; i++) {
        int r = ty + i * 4;
        Wt[(size_t)(n0 + r) * K + k0 + tx] = t[tx][r];
    }
}

// ---------------------------------------------------------------------------
// Bulk fp32 -> bf16 convert (for x). 2048 elems per block.
// ---------------------------------------------------------------------------
__global__ __launch_bounds__(256) void conv_bf16(
    const float* __restrict__ X, unsigned short* __restrict__ Y)
{
    size_t base = (size_t)blockIdx.x * 2048 + (size_t)threadIdx.x * 8;
    float4 a = *(const float4*)(X + base);
    float4 b = *(const float4*)(X + base + 4);
    union { unsigned short u[8]; uint4 q; } o;
    o.u[0] = f2bf(a.x); o.u[1] = f2bf(a.y); o.u[2] = f2bf(a.z); o.u[3] = f2bf(a.w);
    o.u[4] = f2bf(b.x); o.u[5] = f2bf(b.y); o.u[6] = f2bf(b.z); o.u[7] = f2bf(b.w);
    *(uint4*)(Y + base) = o.q;
}

// ---------------------------------------------------------------------------
// GEMM: A(bf16)[M][K] x Wt(bf16)[N][K] + bias (+resid f32)(+relu), *oscale
// Block tile 64(M) x 128(N), 8 waves, wave tile 16x64, K-step 32.
// Double-buffered LDS; async DMA for tile t+1 overlaps WMMA on tile t.
// ---------------------------------------------------------------------------
__global__ __launch_bounds__(256) void gemm_wmma(
    const unsigned short* __restrict__ Abf, const unsigned short* __restrict__ Wt,
    const float* __restrict__ bias, const float* __restrict__ resid,
    float* __restrict__ outf, unsigned short* __restrict__ outb,
    int N, int K, int relu, float oscale)
{
    __shared__ unsigned short lA[2][64 * 32];
    __shared__ unsigned short lB[2][128 * 32];
    const int tid  = threadIdx.x;
    const int lane = tid & 31;
    const int wave = tid >> 5;
    const int wm = wave & 3;
    const int wn = wave >> 2;
    const int m0 = blockIdx.y * 64;
    const int n0 = blockIdx.x * 128;

    FragC acc[4];
#pragma unroll
    for (int i = 0; i < 4; i++)
#pragma unroll
        for (int j = 0; j < 8; j++) acc[i].f[j] = 0.f;

    const int arow   = lane & 15;
    const int kb8    = (lane >> 4) * 8;
    const int half16 = (lane >> 4) * 16;

    const int sr = tid >> 2;            // 0..63
    const int sc = (tid & 3) * 8;       // element offset in K-step
    const unsigned short* gA  = Abf + (size_t)(m0 + sr) * K + sc;
    const unsigned short* gB0 = Wt + (size_t)(n0 + sr) * K + sc;
    const unsigned short* gB1 = Wt + (size_t)(n0 + 64 + sr) * K + sc;

    auto stage = [&](int k0, int bsel) {
#if HAVE_ASYNC
        async_copy16(gA + k0,  ((const uint4*)lA[bsel]) + tid);
        async_copy16(gB0 + k0, ((const uint4*)lB[bsel]) + tid);
        async_copy16(gB1 + k0, ((const uint4*)lB[bsel]) + 256 + tid);
#else
        ((uint4*)lA[bsel])[tid]       = *(const uint4*)(gA + k0);
        ((uint4*)lB[bsel])[tid]       = *(const uint4*)(gB0 + k0);
        ((uint4*)lB[bsel])[tid + 256] = *(const uint4*)(gB1 + k0);
#endif
    };

    stage(0, 0);
    int buf = 0;
    for (int k0 = 0; k0 < K; k0 += 32) {
        const bool more = (k0 + 32) < K;
        if (more) stage(k0 + 32, buf ^ 1);
#if HAVE_ASYNC
        if (more) wait_async<3>(); else wait_async<0>();
#endif
        __syncthreads();

        FragAB a;
        {
            const unsigned short* p = lA[buf] + (wm * 16 + arow) * 32;
            a.q[0] = *(const uint4*)(p + kb8);
            a.q[1] = *(const uint4*)(p + 16 + kb8);
        }
#pragma unroll
        for (int ns = 0; ns < 4; ns++) {
            FragAB bfr;
            const unsigned short* p = lB[buf] + (wn * 64 + ns * 16 + arow) * 32;
            bfr.q[0] = *(const uint4*)(p + kb8);
            bfr.q[1] = *(const uint4*)(p + 16 + kb8);
            acc[ns].v = __builtin_amdgcn_wmma_f32_16x16x32_bf16(
                false, a.v, false, bfr.v, (short)0, acc[ns].v, false, false);
        }
        __syncthreads();
        buf ^= 1;
    }

    const int rbase = m0 + wm * 16 + 8 * (lane >> 4);
#pragma unroll
    for (int ns = 0; ns < 4; ns++) {
        int col = n0 + wn * 64 + ns * 16 + arow;
        float bv = bias[col];
#pragma unroll
        for (int v = 0; v < 8; v++) {
            size_t row = (size_t)(rbase + v);
            float x = acc[ns].f[v] + bv;
            if (resid) x += resid[row * N + col];
            if (relu)  x = x > 0.f ? x : 0.f;
            x *= oscale;
            if (outf) outf[row * N + col] = x;
            if (outb) outb[row * N + col] = f2bf(x);
        }
    }
}

// ---------------------------------------------------------------------------
// Flash-style attention, all-bf16 operands. grid = (SEQ/128, NHEAD, BATCH).
// Q pre-scaled by 1/sqrt(DKH) in its projection GEMM. Double-buffered K/V.
// ---------------------------------------------------------------------------
__global__ __launch_bounds__(256) void attn_wmma(
    const unsigned short* __restrict__ Qb, const unsigned short* __restrict__ Kb,
    const unsigned short* __restrict__ Vb, unsigned short* __restrict__ Ob, int causal)
{
    __shared__ unsigned short lK[2][32 * 64];    // [key][d]
    __shared__ unsigned short lV[2][64 * 32];    // [d][key]
    __shared__ unsigned short lP[8][16 * 32];    // per-wave P scratch

    const int tid = threadIdx.x, lane = tid & 31, wave = tid >> 5;
    const int h = blockIdx.y, b = blockIdx.z;
    const int qb = blockIdx.x * 128 + wave * 16;
    const size_t rowBase = (size_t)b * SEQ;
    const int arow   = lane & 15;
    const int kb8    = (lane >> 4) * 8;
    const int half16 = (lane >> 4) * 16;

    FragAB qf[2];
    {
        const unsigned short* qp = Qb + (rowBase + qb + arow) * EMB + h * DKH;
#pragma unroll
        for (int df = 0; df < 2; df++) {
            qf[df].q[0] = *(const uint4*)(qp + df * 32 + kb8);
            qf[df].q[1] = *(const uint4*)(qp + df * 32 + 16 + kb8);
        }
    }

    float mrow[8], lrow[8];
    FragC oacc[4];
#pragma unroll
    for (int v = 0; v < 8; v++) { mrow[v] = -1e30f; lrow[v] = 0.f; }
#pragma unroll
    for (int i = 0; i < 4; i++)
#pragma unroll
        for (int v = 0; v < 8; v++) oacc[i].f[v] = 0.f;

    const int skey = tid >> 3;         // 0..31
    const int sdc  = (tid & 7) * 8;    // d chunk

    auto stage = [&](int kt, int bsel) {
        const int key0 = kt * 32;
        const unsigned short* gK = Kb + (rowBase + key0 + skey) * EMB + h * DKH + sdc;
#if HAVE_ASYNC
        async_copy16(gK, ((const uint4*)lK[bsel]) + tid);
#else
        ((uint4*)lK[bsel])[tid] = *(const uint4*)gK;
#endif
        union { uint4 q; unsigned short u[8]; } t;   // V^T: b128 load + transpose stores
        t.q = *(const uint4*)(Vb + (rowBase + key0 + skey) * EMB + h * DKH + sdc);
#pragma unroll
        for (int j = 0; j < 8; j++) lV[bsel][(sdc + j) * 32 + skey] = t.u[j];
    };

    const int nkt = causal ? (blockIdx.x * 4 + 4) : (SEQ / 32);
    stage(0, 0);
    int buf = 0;
    for (int kt = 0; kt < nkt; kt++) {
        const int key0 = kt * 32;
        const bool more = (kt + 1) < nkt;
        if (more) stage(kt + 1, buf ^ 1);
#if HAVE_ASYNC
        if (more) wait_async<1>(); else wait_async<0>();
#endif
        __syncthreads();

        if (!causal || key0 <= qb + 15) {   // wave-uniform
            FragC s[2];
#pragma unroll
            for (int ks = 0; ks < 2; ks++) {
#pragma unroll
                for (int v = 0; v < 8; v++) s[ks].f[v] = 0.f;
#pragma unroll
                for (int dc = 0; dc < 2; dc++) {
                    FragAB kf;
                    const unsigned short* p = lK[buf] + (ks * 16 + arow) * 64 + dc * 32;
                    kf.q[0] = *(const uint4*)(p + kb8);
                    kf.q[1] = *(const uint4*)(p + 16 + kb8);
                    s[ks].v = __builtin_amdgcn_wmma_f32_16x16x32_bf16(
                        false, qf[dc].v, false, kf.v, (short)0, s[ks].v, false, false);
                }
            }
            if (causal && key0 + 31 > qb) {
#pragma unroll
                for (int ks = 0; ks < 2; ks++) {
                    int key = key0 + ks * 16 + arow;
#pragma unroll
                    for (int v = 0; v < 8; v++) {
                        int row = qb + v + 8 * (lane >> 4);
                        if (key > row) s[ks].f[v] = -1e30f;
                    }
                }
            }
            float mnew[8], scl[8], psum[8];
#pragma unroll
            for (int v = 0; v < 8; v++) mnew[v] = fmaxf(s[0].f[v], s[1].f[v]);
#pragma unroll
            for (int off = 8; off >= 1; off >>= 1)
#pragma unroll
                for (int v = 0; v < 8; v++)
                    mnew[v] = fmaxf(mnew[v], __shfl_xor(mnew[v], off, 32));
#pragma unroll
            for (int v = 0; v < 8; v++) {
                mnew[v] = fmaxf(mnew[v], mrow[v]);
                scl[v] = __expf(mrow[v] - mnew[v]);
                mrow[v] = mnew[v];
            }
#pragma unroll
            for (int ks = 0; ks < 2; ks++)
#pragma unroll
                for (int v = 0; v < 8; v++)
                    s[ks].f[v] = __expf(s[ks].f[v] - mnew[v]);
#pragma unroll
            for (int v = 0; v < 8; v++) psum[v] = s[0].f[v] + s[1].f[v];
#pragma unroll
            for (int off = 8; off >= 1; off >>= 1)
#pragma unroll
                for (int v = 0; v < 8; v++) psum[v] += __shfl_xor(psum[v], off, 32);
#pragma unroll
            for (int v = 0; v < 8; v++) lrow[v] = lrow[v] * scl[v] + psum[v];

#pragma unroll
            for (int ks = 0; ks < 2; ks++)
#pragma unroll
                for (int v = 0; v < 8; v++)
                    lP[wave][(v + 8 * (lane >> 4)) * 32 + ks * 16 + arow] = f2bf(s[ks].f[v]);
#pragma unroll
            for (int i = 0; i < 4; i++)
#pragma unroll
                for (int v = 0; v < 8; v++) oacc[i].f[v] *= scl[v];

            FragAB pf;
            {
                const unsigned short* p = lP[wave] + arow * 32;
                pf.q[0] = *(const uint4*)(p + kb8);
                pf.q[1] = *(const uint4*)(p + 16 + kb8);
            }
#pragma unroll
            for (int ds4 = 0; ds4 < 4; ds4++) {
                FragAB vf;
                const unsigned short* p = lV[buf] + (ds4 * 16 + arow) * 32;
                vf.q[0] = *(const uint4*)(p + kb8);
                vf.q[1] = *(const uint4*)(p + 16 + kb8);
                oacc[ds4].v = __builtin_amdgcn_wmma_f32_16x16x32_bf16(
                    false, pf.v, false, vf.v, (short)0, oacc[ds4].v, false, false);
            }
        }
        __syncthreads();
        buf ^= 1;
    }

#pragma unroll
    for (int ds4 = 0; ds4 < 4; ds4++) {
        int d = h * DKH + ds4 * 16 + arow;
#pragma unroll
        for (int v = 0; v < 8; v++) {
            int row = qb + v + 8 * (lane >> 4);
            Ob[(rowBase + row) * EMB + d] = f2bf(oacc[ds4].f[v] * (1.0f / lrow[v]));
        }
    }
}

// ---------------------------------------------------------------------------
// LayerNorm over EMB=1024; writes f32 and optional bf16 copy.
// ---------------------------------------------------------------------------
__global__ __launch_bounds__(256) void add_layernorm(
    const float* __restrict__ X, const float* __restrict__ g,
    const float* __restrict__ bt, float* __restrict__ Yf,
    unsigned short* __restrict__ Yb)
{
    __shared__ float red[8];
    const int tid = threadIdx.x, lane = tid & 31, wave = tid >> 5;
    const size_t row = blockIdx.x;
    const float* xr = X + row * EMB;
    float v[4];
    float s = 0.f;
#pragma unroll
    for (int i = 0; i < 4; i++) { v[i] = xr[tid + i * 256]; s += v[i]; }
#pragma unroll
    for (int off = 16; off >= 1; off >>= 1) s += __shfl_xor(s, off, 32);
    if (lane == 0) red[wave] = s;
    __syncthreads();
    float tot = 0.f;
#pragma unroll
    for (int w = 0; w < 8; w++) tot += red[w];
    const float mean = tot * (1.0f / EMB);
    __syncthreads();
    float s2 = 0.f;
#pragma unroll
    for (int i = 0; i < 4; i++) { float d = v[i] - mean; s2 += d * d; }
#pragma unroll
    for (int off = 16; off >= 1; off >>= 1) s2 += __shfl_xor(s2, off, 32);
    if (lane == 0) red[wave] = s2;
    __syncthreads();
    float tv = 0.f;
#pragma unroll
    for (int w = 0; w < 8; w++) tv += red[w];
    const float rstd = rsqrtf(tv * (1.0f / EMB) + LN_EPS);
#pragma unroll
    for (int i = 0; i < 4; i++) {
        int c = tid + i * 256;
        float y = (v[i] - mean) * rstd * g[c] + bt[c];
        Yf[row * EMB + c] = y;
        if (Yb) Yb[row * EMB + c] = f2bf(y);
    }
}

// ---------------------------------------------------------------------------
extern "C" void kernel_launch(void* const* d_in, const int* in_sizes, int n_in,
                              void* d_out, int out_size, void* d_ws, size_t ws_size,
                              hipStream_t stream)
{
    const float* x    = (const float*)d_in[0];
    const float* m_wq = (const float*)d_in[1];  const float* m_bq = (const float*)d_in[2];
    const float* m_wk = (const float*)d_in[3];  const float* m_bk = (const float*)d_in[4];
    const float* m_wv = (const float*)d_in[5];  const float* m_bv = (const float*)d_in[6];
    const float* m_wo = (const float*)d_in[7];  const float* m_bo = (const float*)d_in[8];
    const float* a_wq = (const float*)d_in[9];  const float* a_bq = (const float*)d_in[10];
    const float* a_wk = (const float*)d_in[11]; const float* a_bk = (const float*)d_in[12];
    const float* a_wv = (const float*)d_in[13]; const float* a_bv = (const float*)d_in[14];
    const float* a_wo = (const float*)d_in[15]; const float* a_bo = (const float*)d_in[16];
    const float* f_w1 = (const float*)d_in[17]; const float* f_b1 = (const float*)d_in[18];
    const float* f_w2 = (const float*)d_in[19]; const float* f_b2 = (const float*)d_in[20];
    const float* ln1g = (const float*)d_in[21]; const float* ln1b = (const float*)d_in[22];
    const float* ln2g = (const float*)d_in[23]; const float* ln2b = (const float*)d_in[24];
    const float* ln3g = (const float*)d_in[25]; const float* ln3b = (const float*)d_in[26];

    char* ws = (char*)d_ws;
    size_t off = 0;
    auto alloc = [&](size_t bytes) -> void* {
        void* p = ws + off;
        off += (bytes + 255) & ~(size_t)255;
        return p;
    };
    const size_t WSZ = (size_t)EMB * EMB * sizeof(unsigned short);
    unsigned short* T_mwq = (unsigned short*)alloc(WSZ);
    unsigned short* T_mwk = (unsigned short*)alloc(WSZ);
    unsigned short* T_mwv = (unsigned short*)alloc(WSZ);
    unsigned short* T_mwo = (unsigned short*)alloc(WSZ);
    unsigned short* T_awq = (unsigned short*)alloc(WSZ);
    unsigned short* T_awk = (unsigned short*)alloc(WSZ);
    unsigned short* T_awv = (unsigned short*)alloc(WSZ);
    unsigned short* T_awo = (unsigned short*)alloc(WSZ);
    unsigned short* T_fw1 = (unsigned short*)alloc((size_t)EMB * DFF * sizeof(unsigned short));
    unsigned short* T_fw2 = (unsigned short*)alloc((size_t)DFF * EMB * sizeof(unsigned short));
    const size_t ABSZ = (size_t)TOK * EMB * sizeof(unsigned short);
    const size_t AFSZ = (size_t)TOK * EMB * sizeof(float);
    unsigned short* xbf  = (unsigned short*)alloc(ABSZ);
    unsigned short* qbf  = (unsigned short*)alloc(ABSZ);
    unsigned short* kbf  = (unsigned short*)alloc(ABSZ);
    unsigned short* vbf  = (unsigned short*)alloc(ABSZ);
    unsigned short* aobf = (unsigned short*)alloc(ABSZ);
    unsigned short* h1bf = (unsigned short*)alloc(ABSZ);
    unsigned short* h2bf = (unsigned short*)alloc(ABSZ);
    unsigned short* ffbf = (unsigned short*)alloc((size_t)TOK * DFF * sizeof(unsigned short));
    float* t0 = (float*)alloc(AFSZ);
    float* h1 = (float*)alloc(AFSZ);
    float* h2 = (float*)alloc(AFSZ);

    dim3 tpb(256);
    dim3 gW(EMB / 64, EMB / 64);
    wconv_t<<<gW, tpb, 0, stream>>>(m_wq, T_mwq, EMB, EMB);
    wconv_t<<<gW, tpb, 0, stream>>>(m_wk, T_mwk, EMB, EMB);
    wconv_t<<<gW, tpb, 0, stream>>>(m_wv, T_mwv, EMB, EMB);
    wconv_t<<<gW, tpb, 0, stream>>>(m_wo, T_mwo, EMB, EMB);
    wconv_t<<<gW, tpb, 0, stream>>>(a_wq, T_awq, EMB, EMB);
    wconv_t<<<gW, tpb, 0, stream>>>(a_wk, T_awk, EMB, EMB);
    wconv_t<<<gW, tpb, 0, stream>>>(a_wv, T_awv, EMB, EMB);
    wconv_t<<<gW, tpb, 0, stream>>>(a_wo, T_awo, EMB, EMB);
    wconv_t<<<dim3(EMB / 64, DFF / 64), tpb, 0, stream>>>(f_w1, T_fw1, EMB, DFF);
    wconv_t<<<dim3(DFF / 64, EMB / 64), tpb, 0, stream>>>(f_w2, T_fw2, DFF, EMB);
    conv_bf16<<<(TOK * EMB) / 2048, tpb, 0, stream>>>(x, xbf);

    dim3 gP(EMB / 128, TOK / 64);
    dim3 gF1(DFF / 128, TOK / 64);
    dim3 gA(SEQ / 128, NHEAD, BATCH);

    // ---- block 1: causal self-attention + AddNorm ----
    gemm_wmma<<<gP, tpb, 0, stream>>>(xbf, T_mwq, m_bq, nullptr, nullptr, qbf, EMB, EMB, 0, 0.125f);
    gemm_wmma<<<gP, tpb, 0, stream>>>(xbf, T_mwk, m_bk, nullptr, nullptr, kbf, EMB, EMB, 0, 1.0f);
    gemm_wmma<<<gP, tpb, 0, stream>>>(xbf, T_mwv, m_bv, nullptr, nullptr, vbf, EMB, EMB, 0, 1.0f);
    attn_wmma<<<gA, tpb, 0, stream>>>(qbf, kbf, vbf, aobf, 1);
    gemm_wmma<<<gP, tpb, 0, stream>>>(aobf, T_mwo, m_bo, x, t0, nullptr, EMB, EMB, 0, 1.0f);
    add_layernorm<<<TOK, tpb, 0, stream>>>(t0, ln1g, ln1b, h1, h1bf);

    // ---- block 2: full self-attention + AddNorm ----
    gemm_wmma<<<gP, tpb, 0, stream>>>(h1bf, T_awq, a_bq, nullptr, nullptr, qbf, EMB, EMB, 0, 0.125f);
    gemm_wmma<<<gP, tpb, 0, stream>>>(h1bf, T_awk, a_bk, nullptr, nullptr, kbf, EMB, EMB, 0, 1.0f);
    gemm_wmma<<<gP, tpb, 0, stream>>>(h1bf, T_awv, a_bv, nullptr, nullptr, vbf, EMB, EMB, 0, 1.0f);
    attn_wmma<<<gA, tpb, 0, stream>>>(qbf, kbf, vbf, aobf, 0);
    gemm_wmma<<<gP, tpb, 0, stream>>>(aobf, T_awo, a_bo, h1, t0, nullptr, EMB, EMB, 0, 1.0f);
    add_layernorm<<<TOK, tpb, 0, stream>>>(t0, ln2g, ln2b, h2, h2bf);

    // ---- FFN + AddNorm ----
    gemm_wmma<<<gF1, tpb, 0, stream>>>(h2bf, T_fw1, f_b1, nullptr, nullptr, ffbf, DFF, EMB, 1, 1.0f);
    gemm_wmma<<<gP, tpb, 0, stream>>>(ffbf, T_fw2, f_b2, h2, t0, nullptr, EMB, DFF, 0, 1.0f);
    add_layernorm<<<TOK, tpb, 0, stream>>>(t0, ln3g, ln3b, (float*)d_out, nullptr);
}
